// CombinedUCCModel_22454089023577
// MI455X (gfx1250) — matrix operand order repaired and checked by reading, not compiled, not run on hardware
//
#include <hip/hip_runtime.h>

typedef __attribute__((ext_vector_type(16))) _Float16 v16h;
typedef __attribute__((ext_vector_type(8)))  _Float16 v8h;
typedef __attribute__((ext_vector_type(8)))  float    v8f;

#define AMODE_DENSE  0
#define AMODE_CONV   1   // k=4, s=2, p=1 implicit im2col
#define AMODE_DECONV 2   // transposed conv k=4 s=2 p=1, PARITY-SPLIT: dense 2x2 taps, K=Ci*4
#define OMODE_DENSE  0
#define OMODE_NCHW   1
#define OMODE_NCHW_P2 2  // NCHW scatter into parity class (2y+py, 2x+px) of a 2x upsampled grid
#define ACT_NONE     0
#define ACT_RELU     1
#define ACT_SIGMOID  2

#define HAS_ASYNC_LDS __has_builtin(__builtin_amdgcn_global_load_async_to_lds_b128)
#if HAS_ASYNC_LDS
typedef int v4i __attribute__((vector_size(16)));
typedef __attribute__((address_space(1))) v4i gv4i;   // global-memory 128b chunk
typedef __attribute__((address_space(3))) v4i lv4i;   // LDS 128b chunk
#define ASYNC_WAIT() asm volatile("s_wait_asynccnt 0x0" ::: "memory")
#else
#define ASYNC_WAIT()
#endif

struct GemmDesc {
  const float*     A;     // f32 activations (dense matrix or NCHW tensor)
  const _Float16*  Bt;    // pre-packed f16 [Npad][Kpad] (B transposed)
  const float*     bias;  // f32 [N]
  float*           C;     // f32 output
  int M, N, K, Kpad;      // Kpad % 32 == 0; conv: K=Ci*16, deconv: K=Ci*4
  int amode, omode, act;
  int Ci, IH, IW, OH, OW; // conv geometry; deconv: OH/OW = input grid (quarter-class grid)
  int py, px;             // deconv parity class
};

// ---------------------------------------------------------------------------
// Unified WMMA GEMM: 64x64 C tile per 128-thread block (4 wave32),
// each wave computes 2x2 fragments of 16x16 via v_wmma_f32_16x16x32_f16.
//  - branchless implicit-im2col gather (clamped addrs + select-zero)
//  - spatial offsets/predicates are K-loop invariant (hoisted)
//  - LDS double-buffered: next K-block staging overlaps current WMMAs
//  - B tile staged with GLOBAL_LOAD_ASYNC_TO_LDS (ASYNCcnt-tracked, no VGPR hop)
// ---------------------------------------------------------------------------
__global__ void __launch_bounds__(128) wmma_gemm(GemmDesc d) {
  __shared__ _Float16 Ash[2][64 * 48];  // row stride 48 halfs: 16B-aligned v8h slots
  __shared__ _Float16 Bsh[2][64 * 48];  // stored [col][k]
  const int tid   = threadIdx.x;
  const int lane  = tid & 31;
  const int wv    = tid >> 5;
  const int warpM = wv >> 1;
  const int warpN = wv & 1;
  const int mBase = blockIdx.y * 64;
  const int nBase = blockIdx.x * 64;
  const int HWo   = d.OH * d.OW;
  const int IHW   = d.IH * d.IW;
  const int kshift = (d.amode == AMODE_CONV) ? 4 : 2;  // K-index bits per channel

  v8f acc[2][2] = {};

  const int srow = tid >> 1;        // 0..63 : A row / B column staged by this thread
  const int skl  = (tid & 1) * 16;  // 0 or 16 within the 32-wide K block

  const int  m_g = mBase + srow;
  const bool mok = (m_g < d.M);

  // ---- per-thread, K-loop-invariant im2col precompute ----
  const float* rowbase = d.A;
  int  off32[16];
  bool okq[16];
  if (d.amode == AMODE_DENSE) {
    rowbase = d.A + (size_t)(mok ? m_g : 0) * d.K;
  } else {
    int img = 0, oy = 0, ox = 0;
    if (mok) {
      img = m_g / HWo;
      int rem = m_g - img * HWo;
      oy = rem / d.OW;
      ox = rem - oy * d.OW;
    }
    rowbase = d.A + (size_t)img * d.Ci * IHW;
    #pragma unroll
    for (int q = 0; q < 16; ++q) {
      int iy, ix, cg; bool ok;
      if (d.amode == AMODE_CONV) {            // stride-2, pad-1 conv: (ky,kx)=(q>>2,q&3)
        cg = 0;
        iy = 2 * oy - 1 + (q >> 2);
        ix = 2 * ox - 1 + (q & 3);
      } else {                                 // parity deconv: 2x2 taps (sy,sx)
        cg = q >> 2;                           // channel sub-group within the 32-wide K block
        iy = oy + d.py + ((q >> 1) & 1) - 1;
        ix = ox + d.px + (q & 1) - 1;
      }
      ok = (iy >= 0) & (iy < d.IH) & (ix >= 0) & (ix < d.IW);
      okq[q] = ok & mok;
      int iyc = iy < 0 ? 0 : (iy >= d.IH ? d.IH - 1 : iy);
      int ixc = ix < 0 ? 0 : (ix >= d.IW ? d.IW - 1 : ix);
      off32[q] = cg * IHW + iyc * d.IW + ixc;  // always a valid address
    }
  }

  auto stage = [&](int kb, int buf) {
    // ---- A tile (64 x 32) as f16, branchless gather ----
    _Float16* adst = &Ash[buf][srow * 48 + skl];
    if (d.amode == AMODE_DENSE) {
      int k0 = kb + skl;
      if (mok && k0 < d.K) {
        const float* src = rowbase + k0;
        #pragma unroll
        for (int q = 0; q < 16; ++q) adst[q] = (_Float16)src[q];
      } else {
        #pragma unroll
        for (int q = 0; q < 16; ++q) adst[q] = (_Float16)0.0f;
      }
    } else {
      bool ciok = (kb + skl) < d.K;            // uniform per K-step
      const float* cbase = rowbase + (size_t)(ciok ? ((kb + skl) >> kshift) : 0) * IHW;
      float vals[16];
      #pragma unroll
      for (int q = 0; q < 16; ++q) vals[q] = cbase[off32[q]];  // unconditional loads
      #pragma unroll
      for (int q = 0; q < 16; ++q)
        adst[q] = (okq[q] & ciok) ? (_Float16)vals[q] : (_Float16)0.0f;
    }
    // ---- B tile (64 cols x 32 K), already f16 packed [N][Kpad] ----
    const _Float16* src = d.Bt + (size_t)(nBase + srow) * d.Kpad + kb + skl;
    _Float16* bdst = &Bsh[buf][srow * 48 + skl];
#if HAS_ASYNC_LDS
    __builtin_amdgcn_global_load_async_to_lds_b128((gv4i*)src, (lv4i*)bdst, 0, 0);
    __builtin_amdgcn_global_load_async_to_lds_b128((gv4i*)(src + 8), (lv4i*)(bdst + 8), 0, 0);
#else
    *(v8h*)(bdst)     = *(const v8h*)(src);
    *(v8h*)(bdst + 8) = *(const v8h*)(src + 8);
#endif
    if (kb + 32 < d.Kpad) __builtin_prefetch(src + 32, 0, 0);  // global_prefetch_b8
  };

  stage(0, 0);
  ASYNC_WAIT();
  __syncthreads();

  union Frag { v16h v; v8h h[2]; };
  const int fr  = lane & 15;
  const int ahk = (lane >> 4) << 3;   // A: K pairs 0-7/16-23 (lanes 0-15), 8-15/24-31 (16-31)
  const int bhk = (lane >> 4) << 3;   // B: K 0-7/16-23 (lanes 0-15), 8-15/24-31 (16-31)

  int buf = 0;
  for (int kb = 0; kb < d.Kpad; kb += 32) {
    Frag a[2], b[2];
    #pragma unroll
    for (int i = 0; i < 2; ++i) {
      const _Float16* ap = &Ash[buf][(warpM * 32 + i * 16 + fr) * 48];
      a[i].h[0] = *(const v8h*)(ap + ahk);
      a[i].h[1] = *(const v8h*)(ap + 16 + ahk);
      const _Float16* bp = &Bsh[buf][(warpN * 32 + i * 16 + fr) * 48];
      b[i].h[0] = *(const v8h*)(bp + bhk);
      b[i].h[1] = *(const v8h*)(bp + 16 + bhk);
    }
    if (kb + 32 < d.Kpad) stage(kb + 32, buf ^ 1);  // overlap staging with WMMA
    #pragma unroll
    for (int i = 0; i < 2; ++i)
      #pragma unroll
      for (int j = 0; j < 2; ++j)
        acc[i][j] = __builtin_amdgcn_wmma_f32_16x16x32_f16(
            false, a[i].v, false, b[j].v, (short)0, acc[i][j], false, false);
    ASYNC_WAIT();
    __syncthreads();
    buf ^= 1;
  }

  // ---- epilogue: bias + activation + scatter store ----
  #pragma unroll
  for (int i = 0; i < 2; ++i) {
    #pragma unroll
    for (int j = 0; j < 2; ++j) {
      int col = nBase + warpN * 32 + j * 16 + (lane & 15);
      if (col >= d.N) continue;
      float bv = d.bias[col];
      #pragma unroll
      for (int r = 0; r < 8; ++r) {
        int row = mBase + warpM * 32 + i * 16 + r + ((lane >> 4) << 3);
        if (row >= d.M) continue;
        float v = acc[i][j][r] + bv;
        if (d.act == ACT_RELU)         v = fmaxf(v, 0.0f);
        else if (d.act == ACT_SIGMOID) v = 1.0f / (1.0f + __expf(-v));
        if (d.omode == OMODE_DENSE) {
          d.C[(size_t)row * d.N + col] = v;
        } else if (d.omode == OMODE_NCHW) {
          int im  = row / HWo;
          int rem = row - im * HWo;
          d.C[((size_t)im * d.N + col) * HWo + rem] = v;
        } else {  // OMODE_NCHW_P2: write parity class of 2x-upsampled output
          int im  = row / HWo;
          int rem = row - im * HWo;
          int y   = rem / d.OW;
          int x   = rem - y * d.OW;
          d.C[(((size_t)im * d.N + col) * (2 * d.OH) + 2 * y + d.py) * (2 * d.OW)
              + 2 * x + d.px] = v;
        }
      }
    }
  }
}

// ---------------------------------------------------------------------------
// Weight packing: f32 -> f16, zero-padded [Npad][Kpad]
// ---------------------------------------------------------------------------
__global__ void pack_dense_f16(const float* __restrict__ src, _Float16* __restrict__ dst,
                               int N, int K, int Npad, int Kpad) {
  int idx = blockIdx.x * 256 + threadIdx.x;
  if (idx >= Npad * Kpad) return;
  int n = idx / Kpad, k = idx - n * Kpad;
  float v = (n < N && k < K) ? src[(size_t)n * K + k] : 0.0f;
  dst[idx] = (_Float16)v;
}

// parity-split deconv weights: torch layout w[Ci][Co][4][4]
// Bt[co][ci*4 + sy*2 + sx] = w[ci][co][3-(2*sy+py)][3-(2*sx+px)]
__global__ void pack_deconv_par_f16(const float* __restrict__ w, _Float16* __restrict__ dst,
                                    int Ci, int Co, int Npad, int Kpad, int py, int px) {
  int idx = blockIdx.x * 256 + threadIdx.x;
  if (idx >= Npad * Kpad) return;
  int n = idx / Kpad, k = idx - n * Kpad;
  float v = 0.0f;
  if (n < Co && k < Ci * 4) {
    int ci = k >> 2, sy = (k >> 1) & 1, sx = k & 1;
    int ky = 2 * sy + py, kx = 2 * sx + px;
    v = w[(((size_t)ci * Co + n) * 4 + (3 - ky)) * 4 + (3 - kx)];
  }
  dst[idx] = (_Float16)v;
}

// ---------------------------------------------------------------------------
// BatchNorm (2d and 1d share this: x = [Nimg][C][HW], stats over Nimg*HW)
// ---------------------------------------------------------------------------
__global__ void bn_stats(const float* __restrict__ x, float* __restrict__ stats,
                         int Nimg, int C, int HW, float eps) {
  int c = blockIdx.x;
  __shared__ float ssum[256];
  __shared__ float ssq[256];
  float s = 0.0f, q = 0.0f;
  long total = (long)Nimg * HW;
  for (long i = threadIdx.x; i < total; i += 256) {
    int im  = (int)(i / HW);
    int pos = (int)(i - (long)im * HW);
    float v = x[((size_t)im * C + c) * HW + pos];
    s += v; q += v * v;
  }
  ssum[threadIdx.x] = s; ssq[threadIdx.x] = q;
  __syncthreads();
  for (int st = 128; st > 0; st >>= 1) {
    if (threadIdx.x < st) {
      ssum[threadIdx.x] += ssum[threadIdx.x + st];
      ssq[threadIdx.x]  += ssq[threadIdx.x + st];
    }
    __syncthreads();
  }
  if (threadIdx.x == 0) {
    float mean = ssum[0] / (float)total;
    float var  = ssq[0] / (float)total - mean * mean;
    stats[c * 2 + 0] = mean;
    stats[c * 2 + 1] = rsqrtf(var + eps);
  }
}

__global__ void bn_apply_relu(float* __restrict__ x, const float* __restrict__ stats,
                              const float* __restrict__ g, const float* __restrict__ b,
                              int C, int HW, long total) {
  long i = (long)blockIdx.x * 256 + threadIdx.x;
  if (i >= total) return;
  int c = (int)((i / HW) % C);
  float v = (x[i] - stats[c * 2]) * stats[c * 2 + 1] * g[c] + b[c];
  x[i] = fmaxf(v, 0.0f);
}

// ---------------------------------------------------------------------------
// KDE over bags: enc [B*bag][F] sigmoid activations -> u [B][F][11] normalized
// ---------------------------------------------------------------------------
__global__ void kde_kernel(const float* __restrict__ enc, float* __restrict__ out,
                           int B, int bag, int F) {
  int idx = blockIdx.x * 256 + threadIdx.x;
  if (idx >= B * F) return;
  int b = idx / F, f = idx - b * F;
  const float alpha = 3.9894228040f;  // 1/sqrt(2*pi*sigma^2), sigma=0.1
  const float beta  = -50.0f;         // -1/(2*sigma^2)
  float u[11];
  #pragma unroll
  for (int k = 0; k < 11; ++k) u[k] = 0.0f;
  for (int inst = 0; inst < bag; ++inst) {
    float x = enc[((size_t)b * bag + inst) * F + f];
    #pragma unroll
    for (int k = 0; k < 11; ++k) {
      float dd = x - 0.1f * (float)k;
      u[k] += __expf(beta * dd * dd);
    }
  }
  float s = 0.0f;
  #pragma unroll
  for (int k = 0; k < 11; ++k) { u[k] *= alpha; s += u[k]; }
  float inv = 1.0f / s;
  #pragma unroll
  for (int k = 0; k < 11; ++k) out[((size_t)b * F + f) * 11 + k] = u[k] * inv;
}

// conv1d k=2 s=2, 11->11 channels: z [B][C][2*Lout] -> out [B][C][Lout]
__global__ void conv1d_k2s2(const float* __restrict__ z, const float* __restrict__ w,
                            const float* __restrict__ bias, float* __restrict__ out,
                            int B, int C, int Lout) {
  int idx = blockIdx.x * 256 + threadIdx.x;
  if (idx >= B * C * Lout) return;
  int t = idx % Lout;
  int rem = idx / Lout;
  int o = rem % C;
  int b = rem / C;
  int Lin = Lout * 2;
  float acc = bias[o];
  for (int i = 0; i < C; ++i) {
    const float* zp = z + ((size_t)b * C + i) * Lin + 2 * t;
    const float* wp = w + ((size_t)o * C + i) * 2;
    acc += zp[0] * wp[0] + zp[1] * wp[1];
  }
  out[idx] = acc;
}

// ---------------------------------------------------------------------------
extern "C" void kernel_launch(void* const* d_in, const int* in_sizes, int n_in,
                              void* d_out, int out_size, void* d_ws, size_t ws_size,
                              hipStream_t stream) {
  (void)in_sizes; (void)n_in; (void)out_size; (void)ws_size;
  const float* batch  = (const float*)d_in[0];
  const float* ec1_w  = (const float*)d_in[1];
  const float* ec1_b  = (const float*)d_in[2];
  const float* ebn1_g = (const float*)d_in[3];
  const float* ebn1_b = (const float*)d_in[4];
  const float* ec2_w  = (const float*)d_in[5];
  const float* ec2_b  = (const float*)d_in[6];
  const float* ebn2_g = (const float*)d_in[7];
  const float* ebn2_b = (const float*)d_in[8];
  const float* ec3_w  = (const float*)d_in[9];
  const float* ec3_b  = (const float*)d_in[10];
  const float* ebn3_g = (const float*)d_in[11];
  const float* ebn3_b = (const float*)d_in[12];
  const float* el_w   = (const float*)d_in[13];
  const float* el_b   = (const float*)d_in[14];
  const float* dc1_w  = (const float*)d_in[15];
  const float* dc1_b  = (const float*)d_in[16];
  const float* dc2_w  = (const float*)d_in[17];
  const float* dc2_b  = (const float*)d_in[18];
  const float* dc3_w  = (const float*)d_in[19];
  const float* dc3_b  = (const float*)d_in[20];
  const float* uc1_w  = (const float*)d_in[21];
  const float* uc1_b  = (const float*)d_in[22];
  const float* ubn1_g = (const float*)d_in[23];
  const float* ubn1_b = (const float*)d_in[24];
  const float* uc2_w  = (const float*)d_in[25];
  const float* uc2_b  = (const float*)d_in[26];
  const float* ubn2_g = (const float*)d_in[27];
  const float* ubn2_b = (const float*)d_in[28];
  const float* ul1_w  = (const float*)d_in[29];
  const float* ul1_b  = (const float*)d_in[30];
  const float* ul2_w  = (const float*)d_in[31];
  const float* ul2_b  = (const float*)d_in[32];
  const float* ul3_w  = (const float*)d_in[33];
  const float* ul3_b  = (const float*)d_in[34];
  const float* ul4_w  = (const float*)d_in[35];
  const float* ul4_b  = (const float*)d_in[36];

  float* logits  = (float*)d_out;        // [32,4]
  float* decoded = (float*)d_out + 128;  // [2048,3,32,32]

  char* wsb = (char*)d_ws;
  size_t off = 0;
  auto alloc = [&](size_t bytes) -> void* {
    off = (off + 255) & ~(size_t)255;
    void* p = wsb + off;
    off += bytes;
    return p;
  };
  _Float16* bt_e1  = (_Float16*)alloc((size_t)64 * 64 * 2);
  _Float16* bt_e2  = (_Float16*)alloc((size_t)64 * 512 * 2);
  _Float16* bt_e3  = (_Float16*)alloc((size_t)128 * 1024 * 2);
  _Float16* bt_el  = (_Float16*)alloc((size_t)2048 * 2048 * 2);
  _Float16* bt_d1p = (_Float16*)alloc((size_t)4 * 64 * 512 * 2);  // [parity][Co][Ci*4]
  _Float16* bt_d2p = (_Float16*)alloc((size_t)4 * 64 * 256 * 2);
  _Float16* bt_d3p = (_Float16*)alloc((size_t)4 * 64 * 128 * 2);
  _Float16* bt_u1  = (_Float16*)alloc((size_t)512 * 5632 * 2);
  _Float16* bt_u2  = (_Float16*)alloc((size_t)128 * 512 * 2);
  _Float16* bt_u3  = (_Float16*)alloc((size_t)64 * 128 * 2);
  _Float16* bt_u4  = (_Float16*)alloc((size_t)64 * 32 * 2);
  float* stats = (float*)alloc((size_t)256 * 2 * 4);
  float* y1  = (float*)alloc((size_t)2048 * 32 * 256 * 4);  // enc1 out; reused as deconv2 out
  float* y2  = (float*)alloc((size_t)2048 * 64 * 64 * 4);   // enc2 out; reused as deconv1 out
  float* y3  = (float*)alloc((size_t)2048 * 2048 * 4);      // enc3 out (= linear input)
  float* enc = (float*)alloc((size_t)2048 * 2048 * 4);      // sigmoid embedding
  float* kde = (float*)alloc((size_t)32 * 22528 * 4);
  float* c1  = (float*)alloc((size_t)32 * 11 * 1024 * 4);
  float* c2  = (float*)alloc((size_t)32 * 11 * 512 * 4);
  float* f1  = (float*)alloc((size_t)32 * 512 * 4);
  float* f2  = (float*)alloc((size_t)32 * 128 * 4);
  float* f3  = (float*)alloc((size_t)32 * 32 * 4);
  float* dd1 = y2;
  float* dd2 = y1;

  auto packD = [&](const float* s, _Float16* dst, int N, int K, int Np, int Kp) {
    int total = Np * Kp;
    pack_dense_f16<<<(total + 255) / 256, 256, 0, stream>>>(s, dst, N, K, Np, Kp);
  };
  auto packP = [&](const float* s, _Float16* dst, int Ci, int Co, int Np, int Kp) {
    int total = Np * Kp;
    for (int p = 0; p < 4; ++p)
      pack_deconv_par_f16<<<(total + 255) / 256, 256, 0, stream>>>(
          s, dst + (size_t)p * total, Ci, Co, Np, Kp, p >> 1, p & 1);
  };
  packD(ec1_w, bt_e1, 32, 48, 64, 64);
  packD(ec2_w, bt_e2, 64, 512, 64, 512);
  packD(ec3_w, bt_e3, 128, 1024, 128, 1024);
  packD(el_w,  bt_el, 2048, 2048, 2048, 2048);
  packD(ul1_w, bt_u1, 512, 5632, 512, 5632);
  packD(ul2_w, bt_u2, 128, 512, 128, 512);
  packD(ul3_w, bt_u3, 32, 128, 64, 128);
  packD(ul4_w, bt_u4, 4, 32, 64, 32);
  packP(dc1_w, bt_d1p, 128, 64, 64, 512);
  packP(dc2_w, bt_d2p, 64, 32, 64, 256);
  packP(dc3_w, bt_d3p, 32, 3, 64, 128);

  auto gemm = [&](const float* A, const _Float16* Bt, const float* bias, float* C,
                  int M, int N, int K, int Kpad, int amode, int omode, int act,
                  int Ci, int IH, int IW, int OH, int OW, int py, int px) {
    GemmDesc g;
    g.A = A; g.Bt = Bt; g.bias = bias; g.C = C;
    g.M = M; g.N = N; g.K = K; g.Kpad = Kpad;
    g.amode = amode; g.omode = omode; g.act = act;
    g.Ci = Ci; g.IH = IH; g.IW = IW; g.OH = OH; g.OW = OW;
    g.py = py; g.px = px;
    dim3 grid((N + 63) / 64, (M + 63) / 64);
    wmma_gemm<<<grid, 128, 0, stream>>>(g);
  };
  auto bn = [&](float* x, const float* gg, const float* bb, int Nimg, int C, int HW) {
    bn_stats<<<C, 256, 0, stream>>>(x, stats, Nimg, C, HW, 1e-5f);
    long total = (long)Nimg * C * HW;
    bn_apply_relu<<<(unsigned)((total + 255) / 256), 256, 0, stream>>>(x, stats, gg, bb, C, HW, total);
  };
  // deconv layer = 4 parity-class GEMMs over the quarter-size output grid
  auto deconv = [&](const float* A, const _Float16* Btp, const float* bias, float* C,
                    int Ci, int Co, int IH, int IW, int act) {
    int K = Ci * 4;
    int M = 2048 * IH * IW;
    for (int p = 0; p < 4; ++p)
      gemm(A, Btp + (size_t)p * 64 * K, bias, C, M, Co, K, K,
           AMODE_DECONV, OMODE_NCHW_P2, act, Ci, IH, IW, IH, IW, p >> 1, p & 1);
  };

  // ---- encoder (implicit-im2col WMMA GEMMs) ----
  gemm(batch, bt_e1, ec1_b, y1, 2048 * 256, 32, 48, 64, AMODE_CONV, OMODE_NCHW, ACT_NONE, 3, 32, 32, 16, 16, 0, 0);
  bn(y1, ebn1_g, ebn1_b, 2048, 32, 256);
  gemm(y1, bt_e2, ec2_b, y2, 2048 * 64, 64, 512, 512, AMODE_CONV, OMODE_NCHW, ACT_NONE, 32, 16, 16, 8, 8, 0, 0);
  bn(y2, ebn2_g, ebn2_b, 2048, 64, 64);
  gemm(y2, bt_e3, ec3_b, y3, 2048 * 16, 128, 1024, 1024, AMODE_CONV, OMODE_NCHW, ACT_NONE, 64, 8, 8, 4, 4, 0, 0);
  bn(y3, ebn3_g, ebn3_b, 2048, 128, 16);
  // ---- embedding linear + sigmoid (2048^3 WMMA GEMM) ----
  gemm(y3, bt_el, el_b, enc, 2048, 2048, 2048, 2048, AMODE_DENSE, OMODE_DENSE, ACT_SIGMOID, 0, 0, 0, 1, 1, 0, 0);
  // ---- decoder (parity-split transposed convs, no wasted FLOPs) ----
  deconv(enc, bt_d1p, dc1_b, dd1, 128, 64, 4, 4, ACT_NONE);
  deconv(dd1, bt_d2p, dc2_b, dd2, 64, 32, 8, 8, ACT_NONE);
  deconv(dd2, bt_d3p, dc3_b, decoded, 32, 3, 16, 16, ACT_SIGMOID);
  // ---- UCC head: KDE -> conv1d x2 (+BN/ReLU) -> FC WMMA GEMMs ----
  kde_kernel<<<(32 * 2048 + 255) / 256, 256, 0, stream>>>(enc, kde, 32, 64, 2048);
  conv1d_k2s2<<<(32 * 11 * 1024 + 255) / 256, 256, 0, stream>>>(kde, uc1_w, uc1_b, c1, 32, 11, 1024);
  bn(c1, ubn1_g, ubn1_b, 32, 11, 1024);
  conv1d_k2s2<<<(32 * 11 * 512 + 255) / 256, 256, 0, stream>>>(c1, uc2_w, uc2_b, c2, 32, 11, 512);
  bn(c2, ubn2_g, ubn2_b, 32, 11, 512);
  gemm(c2, bt_u1, ul1_b, f1, 32, 512, 5632, 5632, AMODE_DENSE, OMODE_DENSE, ACT_RELU, 0, 0, 0, 1, 1, 0, 0);
  gemm(f1, bt_u2, ul2_b, f2, 32, 128, 512, 512, AMODE_DENSE, OMODE_DENSE, ACT_RELU, 0, 0, 0, 1, 1, 0, 0);
  gemm(f2, bt_u3, ul3_b, f3, 32, 32, 128, 128, AMODE_DENSE, OMODE_DENSE, ACT_RELU, 0, 0, 0, 1, 1, 0, 0);
  gemm(f3, bt_u4, ul4_b, logits, 32, 4, 32, 32, AMODE_DENSE, OMODE_DENSE, ACT_NONE, 0, 0, 0, 1, 1, 0, 0);
}
